// AttentionModel_14224931684983
// MI455X (gfx1250) — 8-bit matrix operand order repaired and checked by reading, compiled, not run on hardware
//
#include <hip/hip_runtime.h>
#include <cstdint>

typedef __attribute__((ext_vector_type(8)))  int   v8i;
typedef __attribute__((ext_vector_type(16))) int   v16i;
typedef __attribute__((ext_vector_type(8)))  float v8f;

#define B_ 20
#define S_ 4096
#define D_ 64
#define KT 128
#define WAVES 8
#define ROWS_PER_WG (WAVES * 16)

__device__ __forceinline__ uint32_t f32_to_e4m3(float f) {
  uint32_t x = __float_as_uint(f);
  uint32_t sign = (x >> 31) << 7;
  x &= 0x7FFFFFFFu;
  if (x >= 0x43E00000u) return sign | 0x7Eu;
  if (x < 0x3C800000u) {
    float q = rintf(__uint_as_float(x) * 512.0f);
    uint32_t qi = (uint32_t)q;
    return sign | (qi >= 8u ? 0x08u : qi);
  }
  uint32_t r  = x + 0x0007FFFFu + ((x >> 20) & 1u);
  uint32_t em = (r >> 20) - (120u << 3);
  if (em > 0x7Eu) em = 0x7Eu;
  return sign | em;
}

__device__ __forceinline__ uint32_t pack4_fp8(float a, float b, float c, float d) {
#if __has_builtin(__builtin_amdgcn_cvt_pk_fp8_f32)
  int r = 0;
  r = __builtin_amdgcn_cvt_pk_fp8_f32(a, b, r, false);
  r = __builtin_amdgcn_cvt_pk_fp8_f32(c, d, r, true);
  return (uint32_t)r;
#else
  return f32_to_e4m3(a) | (f32_to_e4m3(b) << 8) |
         (f32_to_e4m3(c) << 16) | (f32_to_e4m3(d) << 24);
#endif
}

__device__ __forceinline__ void async_copy16(void* lds, const void* gaddr) {
  const uint32_t lds_off = (uint32_t)(uintptr_t)lds;
  const uint64_t ga      = (uint64_t)(uintptr_t)gaddr;
  asm volatile("global_load_async_to_lds_b128 %0, %1, off"
               :: "v"(lds_off), "v"(ga) : "memory");
}
__device__ __forceinline__ void wait_async0() {
  asm volatile("s_wait_asynccnt 0" ::: "memory");
}

__global__ __launch_bounds__(256) void quant_kv(const float* __restrict__ k,
                                                const float* __restrict__ v,
                                                uint8_t* __restrict__ k8,
                                                uint8_t* __restrict__ v8t) {
  const size_t nK = (size_t)B_ * S_ * D_ / 4;
  size_t t = (size_t)blockIdx.x * blockDim.x + threadIdx.x;
  if (t < nK) {
    float4 f = ((const float4*)k)[t];
    ((uint32_t*)k8)[t] = pack4_fp8(f.x, f.y, f.z, f.w);
  } else if (t < 2 * nK) {
    size_t u = t - nK;
    const int s4 = (int)(u % (S_ / 4));
    const int d  = (int)((u / (S_ / 4)) % D_);
    const int b  = (int)(u / ((size_t)(S_ / 4) * D_));
    const float* vb = v + ((size_t)b * S_ + (size_t)s4 * 4) * D_ + d;
    ((uint32_t*)v8t)[u] = pack4_fp8(vb[0], vb[D_], vb[2 * D_], vb[3 * D_]);
  }
}

__global__ __launch_bounds__(256) void fa_fp8(const float* __restrict__ q,
                                              const uint8_t* __restrict__ k8,
                                              const uint8_t* __restrict__ v8t,
                                              float* __restrict__ out) {
  __shared__ __align__(16) uint8_t smK[2][KT * D_];
  __shared__ __align__(16) uint8_t smV[2][D_ * KT];
  __shared__ __align__(16) uint8_t smP[WAVES][KT * 16];

  const int tid  = threadIdx.x;
  const int wave = tid >> 5;
  const int lane = tid & 31;
  const int m    = lane & 15;
  const int h    = lane >> 4;

  const int b  = blockIdx.x / (S_ / ROWS_PER_WG);
  const int qb = blockIdx.x % (S_ / ROWS_PER_WG);
  const int q0 = qb * ROWS_PER_WG + wave * 16;

  v8i aq;
  {
    const float* qrow = q + ((size_t)b * S_ + q0 + m) * D_;
    #pragma unroll
    for (int v = 0; v < 8; ++v) {
      const int kb = ((v >> 2) << 5) | (((v >> 1) & 1) << 4) | (h << 3) | ((v & 1) << 2);
      float4 f = *(const float4*)(qrow + kb);
      aq[v] = (int)pack4_fp8(f.x, f.y, f.z, f.w);
    }
  }

  v8f acc0 = {}, acc1 = {}, acc2 = {}, acc3 = {};
  float Mrow[8], Lrow[8];
  #pragma unroll
  for (int r = 0; r < 8; ++r) { Mrow[r] = -3.0e38f; Lrow[r] = 0.0f; }

  const float SC = 0.125f * 1.4426950408889634f;

  const uint8_t* gK = k8  + ((size_t)b * S_) * D_;
  const uint8_t* gV = v8t + ((size_t)b * D_) * S_;

  {
    #pragma unroll
    for (int i = 0; i < 2; ++i) {
      const int c = tid + i * 256;
      async_copy16(&smK[0][c * 16], gK + c * 16);
      const int d = c >> 3, part = c & 7;
      async_copy16(&smV[0][c * 16], gV + (size_t)d * S_ + part * 16);
    }
  }

  for (int it = 0; it < S_ / KT; ++it) {
    wait_async0();
    __syncthreads();
    const int buf = it & 1;

    if (it + 1 < S_ / KT) {
      const int kt1 = (it + 1) * KT;
      #pragma unroll
      for (int i = 0; i < 2; ++i) {
        const int c = tid + i * 256;
        async_copy16(&smK[buf ^ 1][c * 16], gK + (size_t)kt1 * D_ + c * 16);
        const int d = c >> 3, part = c & 7;
        async_copy16(&smV[buf ^ 1][c * 16], gV + (size_t)d * S_ + kt1 + part * 16);
      }
    }

    const uint8_t* K = smK[buf];
    const uint8_t* V = smV[buf];
    uint8_t*       P = smP[wave];

    v8f s[8];
    #pragma unroll
    for (int c = 0; c < 8; ++c) {
      v8i bk;
      #pragma unroll
      for (int v = 0; v < 8; ++v) {
        const int db = ((v >> 2) << 5) | (((v >> 1) & 1) << 4) | (h << 3) | ((v & 1) << 2);
        bk[v] = *(const int*)(K + (c * 16 + m) * D_ + db);
      }
      v8f z = {};
      s[c] = __builtin_amdgcn_wmma_f32_16x16x64_fp8_fp8(aq, bk, (short)0, z, false, false);
    }

    float nm[8];
    #pragma unroll
    for (int r = 0; r < 8; ++r) {
      float t0 = s[0][r] * SC, t1 = s[1][r] * SC, t2 = s[2][r] * SC, t3 = s[3][r] * SC;
      float t4 = s[4][r] * SC, t5 = s[5][r] * SC, t6 = s[6][r] * SC, t7 = s[7][r] * SC;
      s[0][r] = t0; s[1][r] = t1; s[2][r] = t2; s[3][r] = t3;
      s[4][r] = t4; s[5][r] = t5; s[6][r] = t6; s[7][r] = t7;
      float mx = fmaxf(fmaxf(fmaxf(t0, t1), fmaxf(t2, t3)),
                       fmaxf(fmaxf(t4, t5), fmaxf(t6, t7)));
      mx = fmaxf(mx, __shfl_xor(mx, 1, 32));
      mx = fmaxf(mx, __shfl_xor(mx, 2, 32));
      mx = fmaxf(mx, __shfl_xor(mx, 4, 32));
      mx = fmaxf(mx, __shfl_xor(mx, 8, 32));
      nm[r] = fmaxf(Mrow[r], mx);
    }
    #pragma unroll
    for (int r = 0; r < 8; ++r) {
      const float corr = exp2f(Mrow[r] - nm[r]);
      Mrow[r] = nm[r];
      acc0[r] *= corr; acc1[r] *= corr; acc2[r] *= corr; acc3[r] *= corr;
      float ps = 0.0f;
      #pragma unroll
      for (int c = 0; c < 8; ++c) {
        float p = exp2f(s[c][r] - nm[r]);
        s[c][r] = p;
        ps += p;
      }
      ps += __shfl_xor(ps, 1, 32);
      ps += __shfl_xor(ps, 2, 32);
      ps += __shfl_xor(ps, 4, 32);
      ps += __shfl_xor(ps, 8, 32);
      Lrow[r] = Lrow[r] * corr + ps;
    }

    #pragma unroll
    for (int c = 0; c < 8; ++c) {
      const uint32_t lo = pack4_fp8(s[c][0], s[c][1], s[c][2], s[c][3]);
      const uint32_t hi = pack4_fp8(s[c][4], s[c][5], s[c][6], s[c][7]);
      const int key = c * 16 + m;
      *(uint32_t*)(P + key * 16 + h * 8)     = lo;
      *(uint32_t*)(P + key * 16 + h * 8 + 4) = hi;
    }
    asm volatile("s_wait_dscnt 0" ::: "memory");

    v16i ap;
    #pragma unroll
    for (int half = 0; half < 2; ++half) {
      #pragma unroll
      for (int v = 0; v < 8; ++v) {
        const int kb = (half << 6) +
            (((v >> 2) << 5) | (((v >> 1) & 1) << 4) | (h << 3) | ((v & 1) << 2));
        uint32_t b0 = P[(kb + 0) * 16 + m];
        uint32_t b1 = P[(kb + 1) * 16 + m];
        uint32_t b2 = P[(kb + 2) * 16 + m];
        uint32_t b3 = P[(kb + 3) * 16 + m];
        ap[half * 8 + v] = (int)(b0 | (b1 << 8) | (b2 << 16) | (b3 << 24));
      }
    }

    #pragma unroll
    for (int j = 0; j < 4; ++j) {
      v16i bv;
      #pragma unroll
      for (int v = 0; v < 16; ++v) {
        const int kb = ((v >> 3) << 6) + ((((v >> 2) & 1) << 5) | (((v >> 1) & 1) << 4) | (h << 3) | ((v & 1) << 2));
        bv[v] = *(const int*)(V + (j * 16 + m) * KT + kb);
      }
      v8f& a = (j == 0 ? acc0 : j == 1 ? acc1 : j == 2 ? acc2 : acc3);
      a = __builtin_amdgcn_wmma_f32_16x16x128_fp8_fp8(ap, bv, (short)0, a, false, false);
    }
  }

  float* orow = out + ((size_t)b * S_ + q0) * D_;
  #pragma unroll
  for (int r = 0; r < 8; ++r) {
    const float inv = 1.0f / Lrow[r];
    const int row = r + 8 * h;
    orow[row * D_ +  0 + m] = acc0[r] * inv;
    orow[row * D_ + 16 + m] = acc1[r] * inv;
    orow[row * D_ + 32 + m] = acc2[r] * inv;
    orow[row * D_ + 48 + m] = acc3[r] * inv;
  }
}

extern "C" void kernel_launch(void* const* d_in, const int* in_sizes, int n_in,
                              void* d_out, int out_size, void* d_ws, size_t ws_size,
                              hipStream_t stream) {
  const float* q = (const float*)d_in[0];
  const float* k = (const float*)d_in[1];
  const float* v = (const float*)d_in[2];
  float* out = (float*)d_out;

  uint8_t* k8  = (uint8_t*)d_ws;
  uint8_t* v8t = k8 + (size_t)B_ * S_ * D_;

  const size_t nq = (size_t)B_ * S_ * D_ / 4 * 2;
  quant_kv<<<(unsigned)((nq + 255) / 256), 256, 0, stream>>>(k, v, k8, v8t);

  fa_fp8<<<B_ * (S_ / ROWS_PER_WG), 256, 0, stream>>>(q, k8, v8t, out);
}
